// DetectionHead_79663053406361
// MI455X (gfx1250) — matrix operand order repaired and checked by reading, not compiled, not run on hardware
//
#include <hip/hip_runtime.h>

// ---------------------------------------------------------------------------
// Detection head: three 1x1 convs as bf16-WMMA GEMMs (fp32 accumulate).
//   out[b,o,p] = sum_c feat[b,c,p] * w[o,c] + bias[o]
// M = out channels (85 -> 96 = 6 tiles), N = spatial (16/wave, 64/WG),
// K = channels, stepped by 32 via v_wmma_f32_16x16x32_bf16.
// ~229 MB HBM traffic -> ~10us @ 23.3 TB/s; bf16 WMMA keeps compute ceiling
// far above the bandwidth roofline (fp32 WMMA x4 would bottleneck).
// This revision: P templated (immediate store/load offsets, no v_mul_u64),
// double-buffered feat staging (1 barrier per K-step), b64 feat loads,
// registers pipelined one K-step ahead of the LDS store.
// ---------------------------------------------------------------------------

typedef __attribute__((ext_vector_type(16))) __bf16 v16bf;
typedef __attribute__((ext_vector_type(8)))  __bf16 v8bf;
typedef __attribute__((ext_vector_type(8)))  float  v8f;
typedef __attribute__((ext_vector_type(4)))  float  v4f;
typedef __attribute__((ext_vector_type(2)))  float  v2f;

#define OUTC   85
#define MPAD   96
#define KCHUNK 256              // channels of A staged in LDS at a time
#define AROW   (KCHUNK + 8)     // bf16 elems per A row; +16B pad -> conflict-free
#define BROW   40               // 32 bf16 + 16B pad per feat row in LDS

template<int C, int P>
__global__ __launch_bounds__(128)
void head1x1_kernel(const float* __restrict__ feat, const float* __restrict__ wgt,
                    const float* __restrict__ bias, float* __restrict__ out)
{
  // 50.7 KB A + 2 x 5 KB B = ~61 KB LDS -> 5 workgroups / WGP.
  __shared__ __attribute__((aligned(32))) __bf16 Alds[MPAD * AROW];
  __shared__ __attribute__((aligned(32))) __bf16 Blds[2][64 * BROW];

  const int t    = threadIdx.x;
  const int b    = blockIdx.y;           // image
  const int n0   = blockIdx.x * 64;      // first spatial column of this block
  const int lane = t & 31;
  const int wv   = t >> 5;               // wave 0..3, 16 columns each
  const int l16  = lane & 15;
  const int h    = lane >> 4;            // half-wave select (WMMA layouts)
  const int n_lane = n0 + wv * 16 + l16;

  // ---- feat staging addressing: each thread owns a column pair + 4 row-pairs
  const int nl  = 2 * (t & 31);                    // local column-pair base
  int col = n0 + nl; if (col > P - 2) col = P - 2; // clamp tail (stores masked)
  const int kr0 = 2 * (t >> 5);                    // row-pair base 0/2/4/6
  const float* fptr = feat + ((size_t)b * C + kr0) * P + col;

  __bf16* const bst[2] = { &Blds[0][nl * BROW + kr0], &Blds[1][nl * BROW + kr0] };
  const __bf16* const bfr[2] = { &Blds[0][(wv * 16 + l16) * BROW + 16 * h],
                                 &Blds[1][(wv * 16 + l16) * BROW + 16 * h] };
  const __bf16* const afr = &Alds[l16 * AROW + 16 * h];

  // ---- accumulators seeded with bias (C operand of the WMMA) ----
  v8f acc[6];
  #pragma unroll
  for (int mt = 0; mt < 6; ++mt) {
    #pragma unroll
    for (int r = 0; r < 8; ++r) {
      int o = mt * 16 + 8 * h + r;
      acc[mt][r] = (o < OUTC) ? bias[o] : 0.0f;
    }
  }

  // ---- pipeline registers: feat block loaded one K-step ahead ----
  v2f ra[4], rb[4];
  auto ldg = [&]() {                       // 8 x global_load_b64, imm offsets
    #pragma unroll
    for (int i = 0; i < 4; ++i) {
      ra[i] = *(const v2f*)(fptr + i * 8 * P);
      rb[i] = *(const v2f*)(fptr + i * 8 * P + P);
    }
    fptr += 32 * P;
  };
  auto stb = [&](__bf16* dst) {            // cvt_pk + ds_store (b32 pairs)
    #pragma unroll
    for (int i = 0; i < 4; ++i) {
      dst[8 * i]            = (__bf16)ra[i][0];   // column nl   : k, k+1
      dst[8 * i + 1]        = (__bf16)rb[i][0];
      dst[8 * i + BROW]     = (__bf16)ra[i][1];   // column nl+1 : k, k+1
      dst[8 * i + BROW + 1] = (__bf16)rb[i][1];
    }
  };
  auto stageA = [&](int kc) {              // W[0:85, kc:kc+256] -> bf16 LDS
    for (int i = 0; i < (MPAD * KCHUNK) / (128 * 4); ++i) {
      int idx = (t + i * 128) * 4;
      int o = idx / KCHUNK;
      int c = idx & (KCHUNK - 1);
      v4f w4 = {0.0f, 0.0f, 0.0f, 0.0f};
      if (o < OUTC) w4 = *(const v4f*)(wgt + (size_t)o * C + kc + c);
      __bf16* dst = &Alds[o * AROW + c];
      dst[0] = (__bf16)w4[0]; dst[1] = (__bf16)w4[1];
      dst[2] = (__bf16)w4[2]; dst[3] = (__bf16)w4[3];
    }
  };

  ldg();                                   // feat block k = 0
  stageA(0);

  for (int kc = 0; kc < C; kc += KCHUNK) {
    if (kc) {                              // A restage between chunks
      __syncthreads();                     // old-A consumers done
      stageA(kc);                          // completion covered by loop barrier
    }
    #pragma unroll
    for (int kk = 0; kk < KCHUNK; kk += 32) {
      const int pbuf = (kk >> 5) & 1;      // compile-time buffer parity
      stb(bst[pbuf]);
      __syncthreads();                     // single barrier per K-step
      if (kk + 32 < KCHUNK || kc + KCHUNK < C) ldg();  // prefetch next block

      // B fragment: lane n holds K = 16h .. 16h+15 contiguous bf16
      const __bf16* bp = bfr[pbuf];
      v8bf bl = *(const v8bf*)bp;
      v8bf bh = *(const v8bf*)(bp + 8);
      v16bf bfrag;
      #pragma unroll
      for (int j = 0; j < 8; ++j) { bfrag[j] = bl[j]; bfrag[j + 8] = bh[j]; }

      // 6 M-tiles: A fragments (K = 16h..16h+15 contiguous) + WMMA
      #pragma unroll
      for (int mt = 0; mt < 6; ++mt) {
        const __bf16* ap = afr + mt * 16 * AROW + kk;
        v8bf al = *(const v8bf*)ap;
        v8bf ah = *(const v8bf*)(ap + 8);
        v16bf afrag;
        #pragma unroll
        for (int j = 0; j < 8; ++j) { afrag[j] = al[j]; afrag[j + 8] = ah[j]; }
        acc[mt] = __builtin_amdgcn_wmma_f32_16x16x32_bf16(
            false, afrag, false, bfrag, (short)0, acc[mt], false, false);
      }
      // buffer pbuf is re-written 2 steps later, fenced by the next barrier
    }
  }

  // ---- store: one base address, all offsets compile-time immediates ----
  if (n_lane < P) {
    float* obase = out + ((size_t)b * OUTC + 8 * h) * P + n_lane;
    #pragma unroll
    for (int mt = 0; mt < 6; ++mt) {
      #pragma unroll
      for (int r = 0; r < 8; ++r) {
        if (mt * 16 + 8 * h + r < OUTC)    // only tile 5 actually masks
          obase[(mt * 16 + r) * P] = acc[mt][r];
      }
    }
  }
}

extern "C" void kernel_launch(void* const* d_in, const int* in_sizes, int n_in,
                              void* d_out, int out_size, void* d_ws, size_t ws_size,
                              hipStream_t stream) {
  (void)in_sizes; (void)n_in; (void)out_size; (void)d_ws; (void)ws_size;
  const float* feat0 = (const float*)d_in[0];
  const float* w0    = (const float*)d_in[1];
  const float* b0    = (const float*)d_in[2];
  const float* feat1 = (const float*)d_in[3];
  const float* w1    = (const float*)d_in[4];
  const float* b1    = (const float*)d_in[5];
  const float* feat2 = (const float*)d_in[6];
  const float* w2    = (const float*)d_in[7];
  const float* b2    = (const float*)d_in[8];

  float* out0 = (float*)d_out;
  float* out1 = out0 + (size_t)16 * 85 * 80 * 80;   // 8,704,000
  float* out2 = out1 + (size_t)16 * 85 * 40 * 40;   // +2,176,000

  // grid.x = ceil(HW/64) spatial blocks, grid.y = batch; 128 threads = 4 waves
  head1x1_kernel<256, 6400><<<dim3(100, 16), 128, 0, stream>>>(feat0, w0, b0, out0);
  head1x1_kernel<512, 1600><<<dim3(25, 16),  128, 0, stream>>>(feat1, w1, b1, out1);
  head1x1_kernel<1024, 400><<<dim3(7, 16),   128, 0, stream>>>(feat2, w2, b2, out2);
}
